// EdgePredictor_HadamandHead_73830487818330
// MI455X (gfx1250) — matrix operand order repaired and checked by reading, not compiled, not run on hardware
//
#include <hip/hip_runtime.h>

// Problem dims (fixed by the reference)
#define T_DIM 16
#define N_DIM 2048
#define D_DIM 256
#define S_DIM 512

typedef __attribute__((ext_vector_type(16))) __bf16 v16bf;
typedef __attribute__((ext_vector_type(8)))  __bf16 v8bf;
typedef __attribute__((ext_vector_type(8)))  float  v8f;

// b128 async-copy payload type per hipcc diagnostic:
//   param 1: '__attribute__((__vector_size__(4 * sizeof(int)))) int __device__ *'
typedef int v4i __attribute__((vector_size(4 * sizeof(int))));
typedef __attribute__((address_space(1))) v4i as1_v4i;   // global
typedef __attribute__((address_space(3))) v4i as3_v4i;   // LDS

// ---------------------------------------------------------------------------
// CDNA5 async Global->LDS copy of 16 bytes (per lane), tracked by ASYNCcnt.
__device__ __forceinline__ void async_cp16(__bf16* lds_dst, const __bf16* gsrc) {
    __builtin_amdgcn_global_load_async_to_lds_b128(
        (as1_v4i*)gsrc, (as3_v4i*)lds_dst, 0, 0);
}

template <int N>
__device__ __forceinline__ void wait_async() {
#if __has_builtin(__builtin_amdgcn_s_wait_asynccnt)
    __builtin_amdgcn_s_wait_asynccnt(N);
#else
    asm volatile("s_wait_asynccnt %0" :: "n"(N));
#endif
}

// ---------------------------------------------------------------------------
// Fragment loaders for V_WMMA_F32_16X16X32_BF16 (wave32), row-major source,
// ld in elements (works for both global and LDS pointers).
//
// A matrix 16x32 bf16 (M x K):
//   lane 0-15  : row M=lane,    V0..V3 = K 0..7,  V4..V7 = K 16..23
//   lane 16-31 : row M=lane-16, V0..V3 = K 8..15, V4..V7 = K 24..31
__device__ __forceinline__ v16bf load_frag_a(const __bf16* src, int ld,
                                             int row0, int k0, int lane) {
    const int row = row0 + (lane & 15);
    const int k   = k0 + ((lane >> 4) << 3);
    const __bf16* p = src + (size_t)row * ld + k;
    union { v16bf v; v8bf h[2]; } u;
    u.h[0] = *reinterpret_cast<const v8bf*>(p);
    u.h[1] = *reinterpret_cast<const v8bf*>(p + 16);
    return u.v;
}

// B matrix 32x16 bf16 (K x N), B_mat[k][n] from row-major src[n][k]:
//   lane 0-15  : col N=lane,    V0..V3 = K 0..7,  V4..V7 = K 16..23
//   lane 16-31 : col N=lane-16, V0..V3 = K 8..15, V4..V7 = K 24..31
__device__ __forceinline__ v16bf load_frag_b(const __bf16* src, int ld,
                                             int n0, int k0, int lane) {
    const int n = n0 + (lane & 15);
    const int k = k0 + ((lane >> 4) << 3);
    const __bf16* p = src + (size_t)n * ld + k;
    union { v16bf v; v8bf h[2]; } u;
    u.h[0] = *reinterpret_cast<const v8bf*>(p);
    u.h[1] = *reinterpret_cast<const v8bf*>(p + 16);
    return u.v;
}

__device__ __forceinline__ v8f wmma_bf16(v16bf a, v16bf b, v8f c) {
    return __builtin_amdgcn_wmma_f32_16x16x32_bf16(
        false, a, false, b, (short)0, c, false, false);
}

// ---------------------------------------------------------------------------
// Kernel 0a: x (f32) -> xbf (bf16), elementwise.
__global__ void cvt_x_kernel(const float* __restrict__ x, __bf16* __restrict__ xbf, int n) {
    int i = blockIdx.x * blockDim.x + threadIdx.x;
    if (i < n) xbf[i] = (__bf16)x[i];
}

// Kernel 0b: W1 [D,S] f32 -> W1^T [S,D] bf16 (so B fragments are contiguous in K).
__global__ void cvt_w1t_kernel(const float* __restrict__ w1, __bf16* __restrict__ w1t) {
    int i = blockIdx.x * blockDim.x + threadIdx.x;
    if (i < D_DIM * S_DIM) {
        int d = i / S_DIM;
        int s = i - d * S_DIM;
        w1t[(size_t)s * D_DIM + d] = (__bf16)w1[i];
    }
}

// ---------------------------------------------------------------------------
// Kernel 1: h = relu(x @ W1 + b1); g = h * W2.  M = T*N = 32768, K = 256, N = 512.
// Block = 128 threads (4 waves); each wave computes a 16x64 strip.
// Direct global fragment loads (small kernel; occupancy hides latency).
__global__ void fc1_kernel(const __bf16* __restrict__ xbf,
                           const __bf16* __restrict__ w1t,
                           const float*  __restrict__ b1,
                           const float*  __restrict__ w2,
                           __bf16* __restrict__ hbf,
                           __bf16* __restrict__ gbf) {
    const int lane  = threadIdx.x & 31;
    const int wave  = threadIdx.x >> 5;
    const int mBase = blockIdx.x * 64 + wave * 16;
    const int nBase = blockIdx.y * 64;

    v8f acc[4] = {};

#pragma unroll
    for (int k0 = 0; k0 < D_DIM; k0 += 32) {
        v16bf a = load_frag_a(xbf, D_DIM, mBase, k0, lane);
#pragma unroll
        for (int c = 0; c < 4; ++c) {
            v16bf b = load_frag_b(w1t, D_DIM, nBase + c * 16, k0, lane);
            acc[c] = wmma_bf16(a, b, acc[c]);
        }
    }

    // C/D layout: VGPR r, lanes 0-15 -> (M=r, N=lane); lanes 16-31 -> (M=r+8, N=lane-16)
#pragma unroll
    for (int c = 0; c < 4; ++c) {
        const int col  = nBase + c * 16 + (lane & 15);
        const float bias = b1[col];
        const float w2v  = w2[col];
#pragma unroll
        for (int r = 0; r < 8; ++r) {
            const int row = mBase + r + ((lane >> 4) << 3);
            float v = acc[c][r] + bias;
            v = v > 0.0f ? v : 0.0f;
            hbf[(size_t)row * S_DIM + col] = (__bf16)v;
            gbf[(size_t)row * S_DIM + col] = (__bf16)(v * w2v);
        }
    }
}

// ---------------------------------------------------------------------------
// Kernel 2: scores[t] = g[t] @ h[t]^T + b2   (2048 x 2048, K = 512) per t.
//
// Block = 256 threads (8 waves) computes a 128x128 C tile.
// K consumed in chunks of 32 staged into LDS by async Global->LDS copies
// (double buffered; ASYNCcnt-pipelined), then each wave computes a 32x64
// sub-tile (2x4 WMMA accumulators) from LDS fragments.
// Grid = (N/128, N/128, T).
__global__ void __launch_bounds__(256)
scores_kernel(const __bf16* __restrict__ hbf,
              const __bf16* __restrict__ gbf,
              const float*  __restrict__ b2,
              float* __restrict__ out) {
    // LDS: per buffer, A chunk 128x32 bf16 (8KB) + B chunk 128x32 (8KB); x2 buffers.
    __shared__ __bf16 ldsA[2][128 * 32];
    __shared__ __bf16 ldsB[2][128 * 32];

    const int tid  = threadIdx.x;
    const int lane = tid & 31;
    const int wave = tid >> 5;
    const int t    = blockIdx.z;
    const int iBlock = blockIdx.y * 128;
    const int jBlock = blockIdx.x * 128;
    const int iLocal = (wave >> 1) * 32;   // 4 waves down
    const int jLocal = (wave & 1) * 64;    // 2 waves across

    const __bf16* __restrict__ h = hbf + (size_t)t * N_DIM * S_DIM;
    const __bf16* __restrict__ g = gbf + (size_t)t * N_DIM * S_DIM;

    // Stage one 32-wide K chunk (kc) of A/B into LDS buffer `buf`.
    // 512 16B pieces per matrix; 256 threads -> 2 pieces each (4 async ops/thread).
    auto stage = [&](int buf, int kc) {
#pragma unroll
        for (int q = 0; q < 2; ++q) {
            const int p   = tid + q * 256;   // piece id, 0..511
            const int row = p >> 2;          // 4 x 16B pieces per 64B row
            const int ko  = p & 3;
            const size_t goff = (size_t)row * S_DIM + kc * 32 + ko * 8;
            async_cp16(&ldsA[buf][p * 8], g + (size_t)iBlock * S_DIM + goff);
            async_cp16(&ldsB[buf][p * 8], h + (size_t)jBlock * S_DIM + goff);
        }
    };

    v8f acc[2][4] = {};

    constexpr int KSTEPS = S_DIM / 32;  // 16
    stage(0, 0);

    for (int kc = 0; kc < KSTEPS; ++kc) {
        const int cur = kc & 1;
        if (kc + 1 < KSTEPS) {
            stage(cur ^ 1, kc + 1);
            wait_async<4>();   // all but the 4 just-issued copies have landed
        } else {
            wait_async<0>();
        }
        __syncthreads();       // chunk `cur` visible to all waves

        const __bf16* cA = ldsA[cur];
        const __bf16* cB = ldsB[cur];
        v16bf af[2], bf[4];
#pragma unroll
        for (int m = 0; m < 2; ++m)
            af[m] = load_frag_a(cA, 32, iLocal + m * 16, 0, lane);
#pragma unroll
        for (int n = 0; n < 4; ++n)
            bf[n] = load_frag_b(cB, 32, jLocal + n * 16, 0, lane);
#pragma unroll
        for (int m = 0; m < 2; ++m)
#pragma unroll
            for (int n = 0; n < 4; ++n)
                acc[m][n] = wmma_bf16(af[m], bf[n], acc[m][n]);

        __syncthreads();       // done reading `cur` before it is overwritten
    }

    const float bias = b2[0];
    float* __restrict__ outT = out + (size_t)t * N_DIM * N_DIM;
#pragma unroll
    for (int m = 0; m < 2; ++m) {
#pragma unroll
        for (int n = 0; n < 4; ++n) {
            const int col = jBlock + jLocal + n * 16 + (lane & 15);
#pragma unroll
            for (int r = 0; r < 8; ++r) {
                const int row = iBlock + iLocal + m * 16 + r + ((lane >> 4) << 3);
                outT[(size_t)row * N_DIM + col] = acc[m][n][r] + bias;
            }
        }
    }
}

// ---------------------------------------------------------------------------
extern "C" void kernel_launch(void* const* d_in, const int* in_sizes, int n_in,
                              void* d_out, int out_size, void* d_ws, size_t ws_size,
                              hipStream_t stream) {
    const float* x  = (const float*)d_in[0];  // [T,N,D]
    const float* W1 = (const float*)d_in[1];  // [D,S]
    const float* b1 = (const float*)d_in[2];  // [S]
    const float* W2 = (const float*)d_in[3];  // [S]
    const float* b2 = (const float*)d_in[4];  // [1]
    float* out = (float*)d_out;               // [T,N,N,1]

    // Workspace layout:
    //   xbf : T*N*D bf16 = 16 MB
    //   w1t : S*D   bf16 = 256 KB
    //   hbf : T*N*S bf16 = 32 MB
    //   gbf : T*N*S bf16 = 32 MB
    char* ws = (char*)d_ws;
    __bf16* xbf = (__bf16*)(ws);
    __bf16* w1t = (__bf16*)(ws + (size_t)16 * 1024 * 1024);
    __bf16* hbf = (__bf16*)(ws + (size_t)16 * 1024 * 1024 + 256 * 1024);
    __bf16* gbf = (__bf16*)(ws + (size_t)48 * 1024 * 1024 + 256 * 1024);

    // 0a) convert x -> bf16
    {
        const int n = T_DIM * N_DIM * D_DIM;  // 8,388,608
        cvt_x_kernel<<<(n + 255) / 256, 256, 0, stream>>>(x, xbf, n);
    }
    // 0b) convert + transpose W1 -> W1^T bf16
    {
        const int n = D_DIM * S_DIM;          // 131,072
        cvt_w1t_kernel<<<(n + 255) / 256, 256, 0, stream>>>(W1, w1t);
    }
    // 1) h = relu(x@W1 + b1); g = h*W2
    {
        dim3 grid((T_DIM * N_DIM) / 64, S_DIM / 64);  // (512, 8)
        fc1_kernel<<<grid, 128, 0, stream>>>(xbf, w1t, b1, W2, hbf, gbf);
    }
    // 2) scores[t] = g[t] @ h[t]^T + b2
    {
        dim3 grid(N_DIM / 128, N_DIM / 128, T_DIM);   // (16, 16, 16)
        scores_kernel<<<grid, 256, 0, stream>>>(hbf, gbf, b2, out);
    }
}
